// HQQLinear_18167711662189
// MI455X (gfx1250) — compile-verified, matrix operand order checked by reading, not run on hardware
//
#include <hip/hip_runtime.h>

typedef __attribute__((ext_vector_type(16))) _Float16 v16h;
typedef __attribute__((ext_vector_type(8)))  _Float16 v8h;
typedef __attribute__((ext_vector_type(4)))  _Float16 v4h;
typedef __attribute__((ext_vector_type(8)))  float    v8f;
typedef __attribute__((ext_vector_type(4)))  int      v4i;
typedef __attribute__((ext_vector_type(8)))  unsigned v8u;

#define OUT_DIM   11008
#define IN_DIM    4096
#define GROUP     64
#define N_GROUPS  (OUT_DIM * IN_DIM / GROUP)   /* 704512 */
#define N_PACKED  (N_GROUPS / 2)               /* 352256 */
#define TOKENS    16
#define HALF_OUT  (OUT_DIM / 2)                /* 5504 */
#define WAVES_PER_BLOCK 4
#define K_PER_WAVE (IN_DIM / WAVES_PER_BLOCK)  /* 1024 */
#define GITER      (K_PER_WAVE / GROUP)        /* 16 group iterations per wave */
#define N_TILES    (HALF_OUT / 16)             /* 344 */

// Dequant core: packed f16 pair (0x6400|nibble = 1024+v) -> one dword holding
// two dequantized f16 weights, via v_fma_mixlo/mixhi_f16 (f32 math, f16 src0).
__device__ __forceinline__ unsigned dq_mix(unsigned hpair, float s, float b) {
    unsigned w;
    asm("v_fma_mixlo_f16 %0, %1, %2, %3 op_sel_hi:[1,0,0]\n\t"
        "v_fma_mixhi_f16 %0, %1, %2, %3 op_sel:[1,0,0] op_sel_hi:[1,0,0]"
        : "=&v"(w)
        : "v"(hpair), "v"(s), "v"(b));
    return w;
}

// One group-iteration's worth of operands (promoted to registers by SROA).
struct Frag {
    v4i   q[8];     // W_q: [ks*4 + c], 8x b128
    v8h   xa[4];    // x fragments: [ks*2 + h], 4x b128
    float s1, z1, s2, z2;
};

__device__ __forceinline__ Frag load_iter(const int* __restrict__ Wq,
                                          const float* __restrict__ scale,
                                          const float* __restrict__ zero,
                                          const _Float16* __restrict__ xr0,
                                          int o1, int kg, int p0) {
    Frag f;
    const int g = o1 * 64 + kg;               // W_q group (high nibble row o1)
    const int* qp = Wq + (long)g * 64;
    const _Float16* xr = xr0 + kg * GROUP;
    #pragma unroll
    for (int ks = 0; ks < 2; ++ks) {
        const int pb = ks * 32 + p0;
        f.q[ks * 4 + 0] = *(const v4i*)(qp + pb);
        f.q[ks * 4 + 1] = *(const v4i*)(qp + pb + 4);
        f.q[ks * 4 + 2] = *(const v4i*)(qp + pb + 16);
        f.q[ks * 4 + 3] = *(const v4i*)(qp + pb + 20);
        f.xa[ks * 2 + 0] = *(const v8h*)(xr + ks * 32 + p0);
        f.xa[ks * 2 + 1] = *(const v8h*)(xr + ks * 32 + p0 + 16);
    }
    f.s1 = scale[g];
    f.z1 = zero[g];
    f.s2 = scale[g + N_PACKED];
    f.z2 = zero[g + N_PACKED];
    return f;
}

__device__ __forceinline__ void compute_iter(const Frag& f, v8f& c1, v8f& c2) {
    const float s1 = f.s1, s2 = f.s2;
    // fold the 0x6400 (=1024+v) f16 bias into the affine constant (f32)
    const float bb1 = -f.z1 * s1 - 1024.0f * s1;
    const float bb2 = -f.z2 * s2 - 1024.0f * s2;

    #pragma unroll
    for (int ks = 0; ks < 2; ++ks) {
        v16h a = __builtin_shufflevector(f.xa[ks * 2], f.xa[ks * 2 + 1],
            0, 1, 2, 3, 4, 5, 6, 7, 8, 9, 10, 11, 12, 13, 14, 15);

        int qq[16];
        *(v4i*)(qq + 0)  = f.q[ks * 4 + 0];
        *(v4i*)(qq + 4)  = f.q[ks * 4 + 1];
        *(v4i*)(qq + 8)  = f.q[ks * 4 + 2];
        *(v4i*)(qq + 12) = f.q[ks * 4 + 3];

        v8u w1u, w2u;                  // packed f16 weights, 2 per dword
        #pragma unroll
        for (int t = 0; t < 8; ++t) {
            const unsigned qa = (unsigned)qq[2 * t];       // byte < 256
            const unsigned qb = (unsigned)qq[2 * t + 1];
            const unsigned pk = (qb << 16) | qa;           // v_lshl_or_b32
            // u16 lanes become f16 values 1024 + nibble (0x6400 | v)
            const unsigned hf = ((pk >> 4) & 0x000F000Fu) | 0x64006400u;
            const unsigned lf = (pk & 0x000F000Fu) | 0x64006400u;
            w1u[t] = dq_mix(hf, s1, bb1);   // high nibble -> tile 1
            w2u[t] = dq_mix(lf, s2, bb2);   // low  nibble -> tile 2
        }
        const v16h w1 = __builtin_bit_cast(v16h, w1u);
        const v16h w2 = __builtin_bit_cast(v16h, w2u);

        c1 = __builtin_amdgcn_wmma_f32_16x16x32_f16(
                 false, a, false, w1, (short)0, c1, false, false);
        c2 = __builtin_amdgcn_wmma_f32_16x16x32_f16(
                 false, a, false, w2, (short)0, c2, false, false);
    }
}

// ---------------------------------------------------------------------------
// Kernel 1: convert x (fp32, row-major [16][4096]) to f16 in workspace.
// ---------------------------------------------------------------------------
__global__ __launch_bounds__(256) void hqq_cvt_x(const float* __restrict__ x,
                                                 _Float16* __restrict__ xh) {
    int i = (blockIdx.x * 256 + threadIdx.x) * 4;   // 65536 elements total
    float4 v = *(const float4*)(x + i);
    v4h o = { (_Float16)v.x, (_Float16)v.y, (_Float16)v.z, (_Float16)v.w };
    *(v4h*)(xh + i) = o;
}

// ---------------------------------------------------------------------------
// Kernel 2: fused dequant + WMMA GEMM, software-pipelined (prefetch depth 1).
// Block owns 16-row tile o_base = blockIdx.x*16 AND its nibble-paired tile
// o_base + 5504 (same W_q bytes: high nibble -> first half, low -> second).
// 4 waves split K (1024 each); LDS reduction split across the waves.
// ---------------------------------------------------------------------------
__global__ __launch_bounds__(128) void hqq_gemm(
    const int*      __restrict__ Wq,     // [N_PACKED][64] packed bytes as int32
    const float*    __restrict__ scale,  // [N_GROUPS]
    const float*    __restrict__ zero,   // [N_GROUPS]
    const float*    __restrict__ bias,   // [OUT_DIM]
    const _Float16* __restrict__ xh,     // [16][4096] f16
    float*          __restrict__ out)    // [16][11008]
{
    __shared__ float red[WAVES_PER_BLOCK * 32 * 16];   // 8 KB

    const int lane = threadIdx.x & 31;
    const int wave = threadIdx.x >> 5;
    const int tile = blockIdx.x;          // 0..343
    const int row  = lane & 15;           // M (token) for A, N (out col) for B
    const int hi   = lane >> 4;           // lane half selects K sub-pattern
    const int p0   = hi * 8;              // K offset within 32-K step

    const int o1  = tile * 16 + row;      // output row, first half (<5504)
    const int kg0 = wave * GITER;         // this wave's first K-group
    const _Float16* xr0 = xh + row * IN_DIM;

    v8f c1 = {};                          // accumulator, rows o1
    v8f c2 = {};                          // accumulator, rows o1 + 5504

    // ---- software pipeline: loads for it+1 overlap compute of it ----------
    Frag f = load_iter(Wq, scale, zero, xr0, o1, kg0, p0);
    for (int it = 0; it < GITER - 1; ++it) {
        Frag fn = load_iter(Wq, scale, zero, xr0, o1, kg0 + it + 1, p0);
        compute_iter(f, c1, c2);
        f = fn;
    }
    compute_iter(f, c1, c2);

    // -------- cross-wave reduction through LDS (rows split across waves) ----
    float* my = red + (wave * 32 + lane) * 16;
    #pragma unroll
    for (int r = 0; r < 8; ++r) { my[r] = c1[r]; my[r + 8] = c2[r]; }
    __syncthreads();

    const int n  = row;                  // D layout: lane%16 = N
    const int oA = tile * 16 + n;
    const int oB = oA + HALF_OUT;
    const float bA = bias[oA];
    const float bB = bias[oB];
    #pragma unroll
    for (int rr = 0; rr < 2; ++rr) {     // each wave handles 2 of 8 rows
        const int r = wave * 2 + rr;
        float sA = 0.f, sB = 0.f;
        #pragma unroll
        for (int w = 0; w < WAVES_PER_BLOCK; ++w) {
            sA += red[(w * 32 + lane) * 16 + r];
            sB += red[(w * 32 + lane) * 16 + r + 8];
        }
        const int m = r + hi * 8;        // D layout: VGPR r -> M=r / r+8
        out[m * OUT_DIM + oA] = sA + bA;
        out[m * OUT_DIM + oB] = sB + bB;
    }
}

// ---------------------------------------------------------------------------
extern "C" void kernel_launch(void* const* d_in, const int* in_sizes, int n_in,
                              void* d_out, int out_size, void* d_ws, size_t ws_size,
                              hipStream_t stream) {
    const float* x     = (const float*)d_in[0];
    const int*   Wq    = (const int*)  d_in[1];
    const float* scale = (const float*)d_in[2];
    const float* zero  = (const float*)d_in[3];
    const float* bias  = (const float*)d_in[4];
    float* out = (float*)d_out;
    _Float16* xh = (_Float16*)d_ws;          // 16*4096*2 = 128 KB

    hqq_cvt_x<<<(TOKENS * IN_DIM) / (256 * 4), 256, 0, stream>>>(x, xh);
    hqq_gemm<<<N_TILES, WAVES_PER_BLOCK * 32, 0, stream>>>(
        Wq, scale, zero, bias, xh, out);
}
